// BasicBlock_51994874085718
// MI455X (gfx1250) — 8-bit matrix operand order repaired and checked by reading, compiled, not run on hardware
//
#include <hip/hip_runtime.h>
#include <hip/hip_bf16.h>

typedef __attribute__((ext_vector_type(16))) int   v16i;
typedef __attribute__((ext_vector_type(8)))  float v8f;

#define N_IMG   64
#define C_CH    256
#define HW      784
#define NPIX    (N_IMG * HW)
#define YELEMS  (N_IMG * C_CH * HW)

#define OFF_SX     0
#define OFF_WB     12845056
#define OFF_SCALE  13434880
#define OFF_SUMS   13435904
#define OFF_SUMSQ  13436928
#define OFF_Y      13437952

__device__ __forceinline__ unsigned fp8sign(float v) {
    return v > 0.f ? 0x38u : (v < 0.f ? 0xB8u : 0u);
}

__global__ void init_stats_kernel(float* sums, float* sumsq) {
    int t = threadIdx.x;
    sums[t]  = 0.f;
    sumsq[t] = 0.f;
}

__global__ void signx_kernel(const float* __restrict__ x, unsigned* __restrict__ sx) {
    int tid = blockIdx.x * 256 + threadIdx.x;
    if (tid >= NPIX * 64) return;
    int c4  = tid & 63;
    int pos = tid >> 6;
    int n = pos / HW, p = pos - n * HW;
    unsigned out = 0;
    #pragma unroll
    for (int j = 0; j < 4; ++j) {
        int c = c4 * 4 + j;
        float v = x[(n * C_CH + c) * HW + p];
        out |= fp8sign(v) << (8 * j);
    }
    sx[pos * 64 + c4] = out;
}

__global__ void scale_kernel(const float* __restrict__ w, float* __restrict__ scale) {
    __shared__ float red[256];
    int o = blockIdx.x;
    const float* wo = w + o * 2304;
    float s = 0.f;
    for (int k = threadIdx.x; k < 2304; k += 256) s += fabsf(wo[k]);
    red[threadIdx.x] = s;
    __syncthreads();
    for (int st = 128; st > 0; st >>= 1) {
        if ((int)threadIdx.x < st) red[threadIdx.x] += red[threadIdx.x + st];
        __syncthreads();
    }
    if (threadIdx.x == 0) scale[o] = red[0] * (1.f / 2304.f);
}

__global__ void packw_kernel(const float* __restrict__ w, unsigned* __restrict__ wb) {
    int idx = blockIdx.x * 256 + threadIdx.x;
    if (idx >= 147456) return;
    int inner = idx & 511;
    int lane  = inner >> 4;
    int vv    = inner & 15;
    int blk   = idx >> 9;
    int otile = blk & 15;
    int sk    = blk >> 4;
    int kb    = sk & 1;
    int s     = sk >> 1;
    int kh = s / 3, kw = s - kh * 3;
    int o  = otile * 16 + (lane & 15);
    int kbase = kb * 128 + (vv >> 1) * 16 + ((lane >= 16) ? 8 : 0) + (vv & 1) * 4;
    unsigned out = 0;
    #pragma unroll
    for (int b = 0; b < 4; ++b) {
        int i = kbase + b;
        float v = w[((o * C_CH + i) * 3 + kh) * 3 + kw];
        out |= fp8sign(v) << (8 * b);
    }
    wb[idx] = out;
}

__global__ __launch_bounds__(256) void conv_wmma_kernel(
        const unsigned char* __restrict__ sx, const unsigned* __restrict__ wb,
        const float* __restrict__ scale, float* __restrict__ y,
        float* __restrict__ sums, float* __restrict__ sumsq) {
    __shared__ int4 ldsB[1024];

    int bid = blockIdx.x;
    int n   = bid / 28;
    int rem = bid - n * 28;
    int pb  = rem >> 2;
    int ob  = rem & 3;

    int lane  = threadIdx.x & 31;
    int wv    = threadIdx.x >> 5;
    int half8 = (lane >> 4) << 3;
    int m     = lane & 15;

    int  p0w  = pb * 128 + wv * 16;
    int  p    = p0w + m;
    int  oh   = p / 28, ow = p - oh * 28;
    bool prow = (p < HW);

    v8f acc[4] = {};

    for (int s = 0; s < 9; ++s) {
        __syncthreads();
        #pragma unroll
        for (int j = 0; j < 4; ++j) {
            int c  = (int)threadIdx.x + j * 256;
            int li = c << 2;
            int kb = li >> 11;
            int ot = (li >> 9) & 3;
            int di = li & 511;
            int g  = ((s * 2 + kb) * 16 + ob * 4 + ot) * 512 + di;
            ldsB[c] = ((const int4*)wb)[g >> 2];
        }
        __syncthreads();

        int  kh = s / 3, kw = s - kh * 3;
        int  ih = oh + kh - 1, iw = ow + kw - 1;
        bool valid = prow && ((unsigned)ih < 28u) && ((unsigned)iw < 28u);
        const char* rowp = (const char*)sx + (((n * 28 + ih) * 28 + iw) << 8);

        #pragma unroll
        for (int kb = 0; kb < 2; ++kb) {
            int2 ch[8] = {};
            if (valid) {
                const int2* ap = (const int2*)(rowp + kb * 128 + half8);
                #pragma unroll
                for (int t = 0; t < 8; ++t) ch[t] = ap[2 * t];
            }
            v16i A = {ch[0].x, ch[0].y, ch[1].x, ch[1].y,
                      ch[2].x, ch[2].y, ch[3].x, ch[3].y,
                      ch[4].x, ch[4].y, ch[5].x, ch[5].y,
                      ch[6].x, ch[6].y, ch[7].x, ch[7].y};
            #pragma unroll
            for (int ot = 0; ot < 4; ++ot) {
                const int4* lb = &ldsB[(kb * 4 + ot) * 128 + lane * 4];
                int4 b0 = lb[0], b1 = lb[1], b2 = lb[2], b3 = lb[3];
                v16i B = {b0.x, b0.y, b0.z, b0.w, b1.x, b1.y, b1.z, b1.w,
                          b2.x, b2.y, b2.z, b2.w, b3.x, b3.y, b3.z, b3.w};
                acc[ot] = __builtin_amdgcn_wmma_f32_16x16x128_fp8_fp8(
                              A, B, (short)0, acc[ot], false, false);
            }
        }
    }

    #pragma unroll
    for (int ot = 0; ot < 4; ++ot) {
        int o = ob * 64 + ot * 16 + m;
        float sc = scale[o];
        float* yo = y + (n * C_CH + o) * HW;
        float sl = 0.f, sq = 0.f;
        #pragma unroll
        for (int v = 0; v < 8; ++v) {
            int pr = p0w + v + half8;
            if (pr < HW) {
                float yv = acc[ot][v] * sc;
                yo[pr] = yv;
                sl += yv;
                sq += yv * yv;
            }
        }
        atomicAdd(&sums[o],  sl);
        atomicAdd(&sumsq[o], sq);
    }
}

__global__ void finalize_kernel(const float* __restrict__ y, const float* __restrict__ x,
                                const float* __restrict__ gamma, const float* __restrict__ beta,
                                const float* __restrict__ sums, const float* __restrict__ sumsq,
                                float* __restrict__ out) {
    int idx = blockIdx.x * 256 + threadIdx.x;
    if (idx >= YELEMS) return;
    int o = (idx / HW) & (C_CH - 1);
    const float invc = 1.f / (float)NPIX;
    float mean = sums[o] * invc;
    float var  = sumsq[o] * invc - mean * mean;
    float inv  = rsqrtf(var + 1e-5f);
    out[idx] = (y[idx] - mean) * inv * gamma[o] + beta[o] + x[idx];
}

extern "C" void kernel_launch(void* const* d_in, const int* in_sizes, int n_in,
                              void* d_out, int out_size, void* d_ws, size_t ws_size,
                              hipStream_t stream) {
    const float* x     = (const float*)d_in[0];
    const float* w     = (const float*)d_in[1];
    const float* gamma = (const float*)d_in[2];
    const float* beta  = (const float*)d_in[3];
    float* out = (float*)d_out;

    char* ws = (char*)d_ws;
    unsigned char* sx = (unsigned char*)(ws + OFF_SX);
    unsigned* wb      = (unsigned*)(ws + OFF_WB);
    float* scale      = (float*)(ws + OFF_SCALE);
    float* sums       = (float*)(ws + OFF_SUMS);
    float* sumsq      = (float*)(ws + OFF_SUMSQ);
    float* y          = (float*)(ws + OFF_Y);

    init_stats_kernel<<<1, 256, 0, stream>>>(sums, sumsq);
    signx_kernel<<<(NPIX * 64) / 256, 256, 0, stream>>>(x, (unsigned*)sx);
    scale_kernel<<<256, 256, 0, stream>>>(w, scale);
    packw_kernel<<<147456 / 256, 256, 0, stream>>>(w, wb);
    conv_wmma_kernel<<<64 * 7 * 4, 256, 0, stream>>>(sx, wb, scale, y, sums, sumsq);
    finalize_kernel<<<YELEMS / 256, 256, 0, stream>>>(y, x, gamma, beta, sums, sumsq, out);
}
